// CausalMultiHeadSelfAttention_35167192219696
// MI455X (gfx1250) — matrix operand order repaired and checked by reading, not compiled, not run on hardware
//
#include <hip/hip_runtime.h>
#include <string.h>

typedef unsigned short ushort_t;
typedef __attribute__((ext_vector_type(16))) __bf16   v16bf;
typedef __attribute__((ext_vector_type(8)))  float    v8f;
typedef unsigned short ushort8 __attribute__((ext_vector_type(8)));
typedef int v4i_vs __attribute__((vector_size(16)));   // matches builtin prototype spelling

#define BATCH 2
#define SEQ   2048
#define DM    1024
#define NH    16
#define HD    64
#define SCALE 0.125f   // 1/sqrt(64)

// ---------- bf16 helpers ----------
__device__ __forceinline__ ushort_t f2bf(float f) {
    unsigned u = __builtin_bit_cast(unsigned, f);
    u += 0x7FFFu + ((u >> 16) & 1u);          // round-to-nearest-even
    return (ushort_t)(u >> 16);
}
__device__ __forceinline__ float bf2f(ushort_t h) {
    unsigned u = ((unsigned)h) << 16;
    return __builtin_bit_cast(float, u);
}

union U16 { v16bf v; ushort8 h[2]; };

// A-fragment (16x32 bf16, K-contiguous rows). base -> row start + k0.
// lanes 0-15 hold K = {0..15}; lanes 16-31 hold K = {16..31}.
__device__ __forceinline__ v16bf load_a_frag(const ushort_t* base, int hi) {
    U16 u;
    u.h[0] = *(const ushort8*)(base + hi * 16);
    u.h[1] = *(const ushort8*)(base + hi * 16 + 8);
    return u.v;
}
// B-fragment (32x16 bf16): per-lane 16 contiguous K values. p already includes hi*16.
__device__ __forceinline__ v16bf load_b_frag(const ushort_t* p) {
    U16 u;
    u.h[0] = *(const ushort8*)(p);
    u.h[1] = *(const ushort8*)(p + 8);
    return u.v;
}

__device__ __forceinline__ v8f wmma_bf16(v16bf a, v16bf b, v8f c) {
    return __builtin_amdgcn_wmma_f32_16x16x32_bf16(false, a, false, b, (short)0, c, false, false);
}

__device__ __forceinline__ float redmax16(float v) {
#pragma unroll
    for (int m = 1; m < 16; m <<= 1) v = fmaxf(v, __shfl_xor(v, m, 32));
    return v;
}
__device__ __forceinline__ float redsum16(float v) {
#pragma unroll
    for (int m = 1; m < 16; m <<= 1) v += __shfl_xor(v, m, 32);
    return v;
}

// ---------- async global->LDS copy of one 16B chunk (per-lane) ----------
__device__ __forceinline__ void async_cp16(const ushort_t* g, ushort_t* l) {
#if defined(__has_builtin) && __has_builtin(__builtin_amdgcn_global_load_async_to_lds_b128)
    __builtin_amdgcn_global_load_async_to_lds_b128(
        (__attribute__((address_space(1))) v4i_vs*)(uintptr_t)g,
        (__attribute__((address_space(3))) v4i_vs*)(unsigned)(uintptr_t)l,
        0, 0);
#else
    *(ushort8*)l = *(const ushort8*)g;     // fallback: VGPR-staged copy
#endif
}
__device__ __forceinline__ void wait_async() {
#if defined(__has_builtin) && __has_builtin(__builtin_amdgcn_s_wait_asynccnt)
    __builtin_amdgcn_s_wait_asynccnt(0);
#else
    asm volatile("s_wait_asynccnt 0" ::: "memory");
#endif
}

// ---------- fp32 -> bf16 conversion ----------
__global__ __launch_bounds__(256) void cvt_bf16_kernel(const float* __restrict__ src,
                                                       ushort_t* __restrict__ dst, int n) {
    int i = blockIdx.x * 256 + threadIdx.x;
    if (i < n) dst[i] = f2bf(src[i]);
}

// ---------- LDS-tiled NT GEMM: C[m,n] = sum_k A[m,k] * B[n,k] ----------
// Block tile 128(M) x 128(N), 8 waves (4x2), each wave 32x64 (2x4 WMMA tiles).
// Double-buffered async k-slices (A 128x32, B 128x32): 8 WMMAs per barrier pair.
// mode 0: bf16 row-major.  mode 1: V head-transpose [b][h][d][s].  mode 2: fp32.
__global__ __launch_bounds__(256) void gemm_nt_kernel(const ushort_t* __restrict__ A,
                                                      const ushort_t* __restrict__ B,
                                                      void* __restrict__ Cout,
                                                      int M, int N, int K, int mode) {
    __shared__ ushort_t As[2][128][32];
    __shared__ ushort_t Bs[2][128][32];

    const int lane    = threadIdx.x;
    const int lane_lo = lane & 15;
    const int hi      = lane >> 4;
    const int w       = threadIdx.y;         // 0..7
    const int wm      = w & 3;                // m sub-tile 0..3 (32 rows each)
    const int wn      = w >> 2;               // n sub-tile 0..1 (64 cols each)
    const int t       = w * 32 + lane;        // 0..255
    const int mBlk    = blockIdx.y * 128;
    const int nBlk    = blockIdx.x * 128;

    const int nk = K / 32;

    // stage k-slice `kk` into buffer `s` (uniform: every thread copies 32B of A and B)
    auto stage = [&](int s, int kk) {
        const ushort_t* ga = A + (size_t)(mBlk + (t >> 1)) * K + kk + (t & 1) * 16;
        ushort_t* la = &As[s][t >> 1][(t & 1) * 16];
        async_cp16(ga, la);
        async_cp16(ga + 8, la + 8);
        const ushort_t* gb = B + (size_t)(nBlk + (t >> 1)) * K + kk + (t & 1) * 16;
        ushort_t* lb = &Bs[s][t >> 1][(t & 1) * 16];
        async_cp16(gb, lb);
        async_cp16(gb + 8, lb + 8);
    };

    v8f acc[2][4] = {{v8f{}, v8f{}, v8f{}, v8f{}}, {v8f{}, v8f{}, v8f{}, v8f{}}};

    stage(0, 0);
    wait_async();
    __syncthreads();

    int buf = 0;
    for (int kt = 0; kt < nk; ++kt) {
        if (kt + 1 < nk) stage(buf ^ 1, (kt + 1) * 32);

        const v16bf a0 = load_a_frag(&As[buf][wm * 32 + lane_lo][0], hi);
        const v16bf a1 = load_a_frag(&As[buf][wm * 32 + 16 + lane_lo][0], hi);
#pragma unroll
        for (int ni = 0; ni < 4; ++ni) {
            const v16bf bn = load_b_frag(&Bs[buf][wn * 64 + ni * 16 + lane_lo][hi * 16]);
            acc[0][ni] = wmma_bf16(a0, bn, acc[0][ni]);
            acc[1][ni] = wmma_bf16(a1, bn, acc[1][ni]);
        }

        wait_async();
        __syncthreads();
        buf ^= 1;
    }

#pragma unroll
    for (int mi = 0; mi < 2; ++mi)
#pragma unroll
        for (int ni = 0; ni < 4; ++ni) {
            const int nn = nBlk + wn * 64 + ni * 16 + lane_lo;
#pragma unroll
            for (int r = 0; r < 8; ++r) {
                const int m = mBlk + wm * 32 + mi * 16 + r + 8 * hi;
                const float val = acc[mi][ni][r];
                if (mode == 0) {
                    ((ushort_t*)Cout)[(size_t)m * N + nn] = f2bf(val);
                } else if (mode == 1) {
                    const int b = m >> 11, s = m & (SEQ - 1);
                    const int h = nn >> 6, d = nn & (HD - 1);
                    ((ushort_t*)Cout)[(((size_t)(b * NH + h)) * HD + d) * SEQ + s] = f2bf(val);
                } else {
                    ((float*)Cout)[(size_t)m * N + nn] = val;
                }
            }
        }
}

// ---------- RoPE on Q and K (bf16, in place) ----------
__global__ __launch_bounds__(256) void rope_kernel(ushort_t* __restrict__ Q,
                                                   ushort_t* __restrict__ Kb,
                                                   const int* __restrict__ tok) {
    int idx = blockIdx.x * 256 + threadIdx.x;            // over BATCH*SEQ*NH*32 pairs
    const int d2 = idx & 31; idx >>= 5;
    const int h  = idx & 15; idx >>= 4;
    const int s  = idx & (SEQ - 1);
    const int b  = idx >> 11;
    if (b >= BATCH) return;

    const float pos = (float)tok[s];
    const float inv = __expf(-((float)(2 * d2) / 64.0f) * 9.2103403719761836f);
    const float ang = pos * inv;
    const float c = __cosf(ang), sn = __sinf(ang);

    const size_t base = (((size_t)b * SEQ + s) * DM) + h * HD + 2 * d2;
#pragma unroll
    for (int which = 0; which < 2; ++which) {
        ushort_t* P = which ? Kb : Q;
        const float e = bf2f(P[base]), o = bf2f(P[base + 1]);
        P[base]     = f2bf(e * c - o * sn);
        P[base + 1] = f2bf(e * sn + o * c);
    }
}

// ---------- flash attention (causal, online softmax, LDS K/V staging) ----------
// grid: (SEQ/64, BATCH*NH), block: (32,4). Each wave: 16 query rows.
// Uniform j-loop per block so all 4 waves cooperatively stage K/V tiles.
__global__ __launch_bounds__(128) void flash_kernel(const ushort_t* __restrict__ Qb,
                                                    const ushort_t* __restrict__ Kb,
                                                    const ushort_t* __restrict__ Vt,
                                                    ushort_t* __restrict__ attnb) {
    __shared__ ushort_t Ks[32][64];       // [key row 0..31][d 0..63]
    __shared__ ushort_t Vs[64][32];       // [d 0..63][key row 0..31]
    __shared__ ushort_t plds[4][16][40];  // per-wave P tile (padded rows, 16B aligned)

    const int bh = blockIdx.y;
    const int b = bh >> 4, h = bh & 15;
    const int w       = threadIdx.y;
    const int lane    = threadIdx.x;
    const int lane_lo = lane & 15;
    const int hi      = lane >> 4;
    const int t       = w * 32 + lane;            // 0..127
    const int q0      = blockIdx.x * 64 + w * 16;
    const int jmax    = blockIdx.x * 64 + 63;

    const ushort_t* Qr = Qb + ((size_t)b * SEQ) * DM + h * HD;
    const ushort_t* Kr = Kb + ((size_t)b * SEQ) * DM + h * HD;
    const ushort_t* Vh = Vt + ((size_t)(b * NH + h)) * HD * SEQ;   // [d][s]

    // Q fragments (rows q0..q0+15, d = 0..63), held for the whole kernel
    const ushort_t* qptr = Qr + (size_t)(q0 + lane_lo) * DM;
    const v16bf qa0 = load_a_frag(qptr, hi);        // d 0..31
    const v16bf qa1 = load_a_frag(qptr + 32, hi);   // d 32..63

    v8f acc[4] = {v8f{}, v8f{}, v8f{}, v8f{}};
    float m_run[8], l_run[8];
#pragma unroll
    for (int r = 0; r < 8; ++r) { m_run[r] = -1e30f; l_run[r] = 0.0f; }

    for (int j0 = 0; j0 <= jmax; j0 += 32) {
        // ---- cooperative stage of K tile (32x64) and Vt tile (64x32) ----
        __syncthreads();    // previous iteration done reading Ks/Vs
        {
            const ushort_t* gk = Kr + (size_t)(j0 + (t >> 2)) * DM + (t & 3) * 16;
            ushort_t* lk = &Ks[t >> 2][(t & 3) * 16];
            async_cp16(gk, lk);
            async_cp16(gk + 8, lk + 8);
            const ushort_t* gv = Vh + (size_t)(t >> 1) * SEQ + j0 + (t & 1) * 16;
            ushort_t* lv = &Vs[t >> 1][(t & 1) * 16];
            async_cp16(gv, lv);
            async_cp16(gv + 8, lv + 8);
        }
        wait_async();
        __syncthreads();

        // ---- scores: two 16x16 tiles (cols j0..+15 and j0+16..+31) ----
        v8f s_lo = {}, s_hi = {};
        s_lo = wmma_bf16(qa0, load_b_frag(&Ks[lane_lo][hi * 16]), s_lo);
        s_lo = wmma_bf16(qa1, load_b_frag(&Ks[lane_lo][32 + hi * 16]), s_lo);
        s_hi = wmma_bf16(qa0, load_b_frag(&Ks[16 + lane_lo][hi * 16]), s_hi);
        s_hi = wmma_bf16(qa1, load_b_frag(&Ks[16 + lane_lo][32 + hi * 16]), s_hi);

        // ---- online softmax (per-row, rows r+8*hi) ----
        float alpha_[8];
#pragma unroll
        for (int r = 0; r < 8; ++r) {
            const int rowi = q0 + r + 8 * hi;
            const int c_lo = j0 + lane_lo, c_hi = j0 + 16 + lane_lo;
            float v0 = (c_lo <= rowi) ? s_lo[r] * SCALE : -1e30f;
            float v1 = (c_hi <= rowi) ? s_hi[r] * SCALE : -1e30f;
            const float mt  = redmax16(fmaxf(v0, v1));
            const float mn  = fmaxf(m_run[r], mt);
            const float al  = __expf(m_run[r] - mn);
            const float p0  = __expf(v0 - mn);
            const float p1  = __expf(v1 - mn);
            l_run[r] = l_run[r] * al + redsum16(p0 + p1);
            m_run[r] = mn;
            alpha_[r] = al;
            s_lo[r] = p0; s_hi[r] = p1;
        }
#pragma unroll
        for (int tt = 0; tt < 4; ++tt)
#pragma unroll
            for (int r = 0; r < 8; ++r) acc[tt][r] *= alpha_[r];

        // ---- stage P tile (16x32 bf16) in wave-private LDS, re-read as A-frag ----
#pragma unroll
        for (int r = 0; r < 8; ++r) {
            plds[w][r + 8 * hi][lane_lo]      = f2bf(s_lo[r]);
            plds[w][r + 8 * hi][16 + lane_lo] = f2bf(s_hi[r]);
        }
        asm volatile("s_wait_dscnt 0" ::: "memory");
        U16 pu;
        pu.h[0] = *(const ushort8*)(&plds[w][lane_lo][hi * 16]);
        pu.h[1] = *(const ushort8*)(&plds[w][lane_lo][hi * 16 + 8]);
        const v16bf pa = pu.v;

        // ---- P (16x32) x V (32x64): 4 WMMAs over d tiles ----
#pragma unroll
        for (int tt = 0; tt < 4; ++tt)
            acc[tt] = wmma_bf16(pa, load_b_frag(&Vs[tt * 16 + lane_lo][hi * 16]), acc[tt]);
    }

    // ---- epilogue: normalize, store bf16 attn (B,S,DM layout) ----
#pragma unroll
    for (int r = 0; r < 8; ++r) {
        const float inv = 1.0f / l_run[r];
        const size_t row = ((size_t)b * SEQ + (q0 + r + 8 * hi)) * DM + h * HD;
#pragma unroll
        for (int tt = 0; tt < 4; ++tt)
            attnb[row + tt * 16 + lane_lo] = f2bf(acc[tt][r] * inv);
    }
}

// ---------- host launcher ----------
extern "C" void kernel_launch(void* const* d_in, const int* in_sizes, int n_in,
                              void* d_out, int out_size, void* d_ws, size_t ws_size,
                              hipStream_t stream) {
    const float* x  = (const float*)d_in[0];
    const int* tok  = (const int*)d_in[1];
    const float* Wq = (const float*)d_in[2];
    const float* Wk = (const float*)d_in[3];
    const float* Wv = (const float*)d_in[4];
    const float* Wo = (const float*)d_in[5];

    char* ws = (char*)d_ws;
    const size_t XB   = 0;                      // x bf16       8 MiB
    const size_t WQB  = XB  + (8u << 20);       // 2 MiB each
    const size_t WKB  = WQB + (2u << 20);
    const size_t WVB  = WKB + (2u << 20);
    const size_t WOB  = WVB + (2u << 20);
    const size_t QB   = WOB + (2u << 20);       // 8 MiB
    const size_t KB   = QB  + (8u << 20);       // 8 MiB
    const size_t VT   = KB  + (8u << 20);       // 8 MiB
    const size_t ATT  = VT  + (8u << 20);       // 8 MiB

    ushort_t* xb  = (ushort_t*)(ws + XB);
    ushort_t* wqb = (ushort_t*)(ws + WQB);
    ushort_t* wkb = (ushort_t*)(ws + WKB);
    ushort_t* wvb = (ushort_t*)(ws + WVB);
    ushort_t* wob = (ushort_t*)(ws + WOB);
    ushort_t* qb  = (ushort_t*)(ws + QB);
    ushort_t* kb  = (ushort_t*)(ws + KB);
    ushort_t* vt  = (ushort_t*)(ws + VT);
    ushort_t* att = (ushort_t*)(ws + ATT);

    const int NX = BATCH * SEQ * DM;   // 4194304
    const int NW = DM * DM;            // 1048576

    cvt_bf16_kernel<<<NX / 256, 256, 0, stream>>>(x,  xb,  NX);
    cvt_bf16_kernel<<<NW / 256, 256, 0, stream>>>(Wq, wqb, NW);
    cvt_bf16_kernel<<<NW / 256, 256, 0, stream>>>(Wk, wkb, NW);
    cvt_bf16_kernel<<<NW / 256, 256, 0, stream>>>(Wv, wvb, NW);
    cvt_bf16_kernel<<<NW / 256, 256, 0, stream>>>(Wo, wob, NW);

    const int M = BATCH * SEQ;         // 4096
    dim3 gblk(32, 8);
    dim3 ggrd(DM / 128, M / 128);      // (8, 32)
    gemm_nt_kernel<<<ggrd, gblk, 0, stream>>>(xb, wqb, qb,  M, DM, DM, 0);
    gemm_nt_kernel<<<ggrd, gblk, 0, stream>>>(xb, wkb, kb,  M, DM, DM, 0);
    gemm_nt_kernel<<<ggrd, gblk, 0, stream>>>(xb, wvb, vt,  M, DM, DM, 1);

    const int NPAIR = BATCH * SEQ * DM / 2;    // 2097152
    rope_kernel<<<NPAIR / 256, 256, 0, stream>>>(qb, kb, tok);

    dim3 fblk(32, 4);
    dim3 fgrd(SEQ / 64, BATCH * NH);   // (32, 32)
    flash_kernel<<<fgrd, fblk, 0, stream>>>(qb, kb, vt, att);

    gemm_nt_kernel<<<ggrd, gblk, 0, stream>>>(att, wob, d_out, M, DM, DM, 2);
}
